// Mamba_Auto_Regression_86019605004373
// MI455X (gfx1250) — matrix operand order repaired and checked by reading, not compiled, not run on hardware
//
#include <hip/hip_runtime.h>
#include <stdint.h>

#define DMODEL 128
#define DINNER 256
#define LSEQ   8192
#define NBATCH 4
#define NPTS   (NBATCH*LSEQ)
#define HGRD   512
#define WGRD   512

typedef __bf16 bf16x16 __attribute__((ext_vector_type(16)));
typedef float  f32x8   __attribute__((ext_vector_type(8)));

union Frag { bf16x16 v; unsigned int u[8]; };

__device__ __forceinline__ unsigned short f2bf(float f) {
  union { float f; unsigned int u; } x; x.f = f;
  unsigned int r = x.u + 0x7fffu + ((x.u >> 16) & 1u);
  return (unsigned short)(r >> 16);
}

// pack two f32 -> packed bf16 pair; prefer hardware v_cvt_pk_bf16_f32 when available
template<int DUMMY = 0>
__device__ __forceinline__ unsigned int pack2(float a, float b) {
#if __has_builtin(__builtin_amdgcn_cvt_pk_bf16_f32)
  using pk_t = decltype(__builtin_amdgcn_cvt_pk_bf16_f32(0.0f, 0.0f));
  if constexpr (sizeof(pk_t) == 4) {
    return __builtin_bit_cast(unsigned int, __builtin_amdgcn_cvt_pk_bf16_f32(a, b));
  } else {
    return (unsigned int)f2bf(a) | ((unsigned int)f2bf(b) << 16);
  }
#else
  return (unsigned int)f2bf(a) | ((unsigned int)f2bf(b) << 16);
#endif
}

__device__ __forceinline__ void packA(Frag& a, int base, float4 lo, float4 hi) {
  a.u[base+0] = pack2(lo.x, lo.y);
  a.u[base+1] = pack2(lo.z, lo.w);
  a.u[base+2] = pack2(hi.x, hi.y);
  a.u[base+3] = pack2(hi.z, hi.w);
}
__device__ __forceinline__ void loadB(Frag& f, const unsigned short* W) {
  const uint4* wp = (const uint4*)W;
  uint4 w0 = wp[0], w1 = wp[1];
  f.u[0]=w0.x; f.u[1]=w0.y; f.u[2]=w0.z; f.u[3]=w0.w;
  f.u[4]=w1.x; f.u[5]=w1.y; f.u[6]=w1.z; f.u[7]=w1.w;
}
__device__ __forceinline__ f32x8 wmma_bf16(const Frag& a, const Frag& b, f32x8 c) {
  return __builtin_amdgcn_wmma_f32_16x16x32_bf16(false, a.v, false, b.v, (short)0, c, false, false);
}
__device__ __forceinline__ float silu(float z) {
  return z / (1.0f + __expf(-z));
}

// ---------------------------------------------------------------- misc prep
__global__ void k_init_vol(int* __restrict__ vol) {
  int gid = blockIdx.x * 256 + threadIdx.x;
  vol[gid] = -1;
}
__global__ void k_scatter_vol(const int* __restrict__ coords, int* __restrict__ vol) {
  int gid = blockIdx.x * 256 + threadIdx.x;
  int b = coords[gid*3], y = coords[gid*3+1], x = coords[gid*3+2];
  vol[((size_t)b*HGRD + y)*WGRD + x] = gid;
}
__global__ void k_f2bf(const float* __restrict__ src, unsigned short* __restrict__ dst) {
  int gid = blockIdx.x * 256 + threadIdx.x;
  dst[gid] = f2bf(src[gid]);
}
// xproj (40x256) -> padded (48x256), pad rows are zero
__global__ void k_f2bf_pad(const float* __restrict__ src, unsigned short* __restrict__ dst) {
  int gid = blockIdx.x * 256 + threadIdx.x;  // 48*256
  int r = gid >> 8;
  dst[gid] = (r < 40) ? f2bf(src[gid]) : (unsigned short)0;
}
// conv weights (9)[k=128][n=128] -> bf16 [tap][n][k]
__global__ void k_convw2bf(const float* __restrict__ src, unsigned short* __restrict__ dst) {
  int gid = blockIdx.x * 256 + threadIdx.x;  // 9*128*128
  int t = gid >> 14;
  int n = (gid >> 7) & 127;
  int k = gid & 127;
  dst[gid] = f2bf(src[(t << 14) + (k << 7) + n]);
}

// ---------------------------------------------------------------- generic WMMA GEMM
// C[M x Ncols] = scale(A) * W^T, W stored bf16 [Ncols][K] row-major.
// Optional per-row rscale (RMSNorm) and per-col gcol (norm gamma) folded into A.
// flip=1 remaps A row (b,l) -> (b, L-1-l) for the backward Mamba branch.
// K is compile-time: full unroll of the accumulation chain.
template<int NT, int K>
__global__ void k_gemm(const float* __restrict__ A, int lda,
                       const float* __restrict__ rscale,
                       const float* __restrict__ gcol,
                       const unsigned short* __restrict__ W,
                       float* __restrict__ C, int ldc, int flip)
{
  const int lane = threadIdx.x & 31;
  const int wave = threadIdx.x >> 5;
  const int l15  = lane & 15;
  const int lh   = lane >> 4;
  const int row0 = (blockIdx.x * 4 + wave) * 16;
  const int colbase = blockIdx.y * (16 * NT);

  int orow = row0 + l15;
  int arow = orow;
  if (flip) { int b = orow >> 13; int l = orow & (LSEQ-1); arow = (b << 13) + (LSEQ-1-l); }
  const float* ap = A + (size_t)arow * lda;
  const float s = rscale ? rscale[arow] : 1.0f;

  f32x8 acc[NT];
#pragma unroll
  for (int t = 0; t < NT; ++t) acc[t] = (f32x8)0.0f;

#pragma unroll
  for (int kc = 0; kc < K; kc += 32) {
    Frag a;
    {
      int k0 = kc + lh*16;
      float4 x0 = *(const float4*)(ap + k0);
      float4 x1 = *(const float4*)(ap + k0 + 4);
      int k1 = kc + lh*16 + 8;
      float4 x2 = *(const float4*)(ap + k1);
      float4 x3 = *(const float4*)(ap + k1 + 4);
      if (rscale) {
        x0.x*=s; x0.y*=s; x0.z*=s; x0.w*=s; x1.x*=s; x1.y*=s; x1.z*=s; x1.w*=s;
        x2.x*=s; x2.y*=s; x2.z*=s; x2.w*=s; x3.x*=s; x3.y*=s; x3.z*=s; x3.w*=s;
      }
      if (gcol) {
        float4 g0 = *(const float4*)(gcol + k0);
        float4 g1 = *(const float4*)(gcol + k0 + 4);
        float4 g2 = *(const float4*)(gcol + k1);
        float4 g3 = *(const float4*)(gcol + k1 + 4);
        x0.x*=g0.x; x0.y*=g0.y; x0.z*=g0.z; x0.w*=g0.w;
        x1.x*=g1.x; x1.y*=g1.y; x1.z*=g1.z; x1.w*=g1.w;
        x2.x*=g2.x; x2.y*=g2.y; x2.z*=g2.z; x2.w*=g2.w;
        x3.x*=g3.x; x3.y*=g3.y; x3.z*=g3.z; x3.w*=g3.w;
      }
      packA(a, 0, x0, x1);
      packA(a, 4, x2, x3);
    }
    int kb = kc + lh*16;
#pragma unroll
    for (int t = 0; t < NT; ++t) {
      int n = colbase + t*16 + l15;
      Frag b;
      loadB(b, W + (size_t)n*K + kb);
      acc[t] = wmma_bf16(a, b, acc[t]);
    }
  }
#pragma unroll
  for (int t = 0; t < NT; ++t)
#pragma unroll
    for (int r = 0; r < 8; ++r)
      C[(size_t)(row0 + r + lh*8) * ldc + colbase + t*16 + l15] = acc[t][r];
}

// ---------------------------------------------------------------- sparse 3x3 conv (gathered WMMA GEMM)
__global__ void k_conv(const float* __restrict__ Fin, const int* __restrict__ coords,
                       const int* __restrict__ vol, const unsigned short* __restrict__ W,
                       float* __restrict__ C)
{
  const int lane = threadIdx.x & 31;
  const int wave = threadIdx.x >> 5;
  const int l15  = lane & 15;
  const int lh   = lane >> 4;
  const int row0 = (blockIdx.x * 4 + wave) * 16;
  const int colbase = blockIdx.y * 64;

  const int orow = row0 + l15;
  const int b = coords[orow*3], y = coords[orow*3+1], x = coords[orow*3+2];

  f32x8 acc[4];
#pragma unroll
  for (int t = 0; t < 4; ++t) acc[t] = (f32x8)0.0f;

  for (int tap = 0; tap < 9; ++tap) {
    int dy = tap/3 - 1, dx = tap%3 - 1;
    int ny = y + dy, nx = x + dx;
    int nb = -1;
    if (ny >= 0 && ny < HGRD && nx >= 0 && nx < WGRD)
      nb = vol[((size_t)b*HGRD + ny)*WGRD + nx];
    const float* ap = Fin + (size_t)(nb < 0 ? 0 : nb) * DMODEL;
    const unsigned short* wt = W + (size_t)tap * DMODEL * DMODEL;

#pragma unroll
    for (int kc = 0; kc < DMODEL; kc += 32) {
      Frag a;
      if (nb >= 0) {
        int k0 = kc + lh*16;
        float4 x0 = *(const float4*)(ap + k0);
        float4 x1 = *(const float4*)(ap + k0 + 4);
        int k1 = kc + lh*16 + 8;
        float4 x2 = *(const float4*)(ap + k1);
        float4 x3 = *(const float4*)(ap + k1 + 4);
        packA(a, 0, x0, x1);
        packA(a, 4, x2, x3);
      } else {
#pragma unroll
        for (int i = 0; i < 8; ++i) a.u[i] = 0u;
      }
      int kb = kc + lh*16;
#pragma unroll
      for (int t = 0; t < 4; ++t) {
        int n = colbase + t*16 + l15;
        Frag bf;
        loadB(bf, wt + (size_t)n*DMODEL + kb);
        acc[t] = wmma_bf16(a, bf, acc[t]);
      }
    }
  }
#pragma unroll
  for (int t = 0; t < 4; ++t)
#pragma unroll
    for (int r = 0; r < 8; ++r)
      C[(size_t)(row0 + r + lh*8) * DMODEL + colbase + t*16 + l15] = acc[t][r];
}

// ---------------------------------------------------------------- enc BN+ReLU + positional MLP, in place on h
__global__ void k_pe_add(float* __restrict__ h, const int* __restrict__ coords,
                         const float* __restrict__ pw1, const float* __restrict__ pb1,
                         const float* __restrict__ pw2, const float* __restrict__ pb2,
                         const float* __restrict__ pg, const float* __restrict__ pb,
                         const float* __restrict__ pm, const float* __restrict__ pv,
                         const float* __restrict__ eg, const float* __restrict__ eb,
                         const float* __restrict__ em, const float* __restrict__ ev)
{
  __shared__ float sh[DMODEL];
  __shared__ float p8[8];
  const int pos = blockIdx.x;
  const int o = threadIdx.x;
  if (o == 0) {
    int y = coords[pos*3+1], x = coords[pos*3+2];
    p8[0] = (float)(y/12);     p8[1] = (float)(x/12);
    p8[2] = (float)(y%12);     p8[3] = (float)(x%12);
    p8[4] = (float)((y+6)/12); p8[5] = (float)((x+6)/12);
    p8[6] = (float)((y+6)%12); p8[7] = (float)((x+6)%12);
  }
  __syncthreads();
  float s = pb1[o];
#pragma unroll
  for (int i = 0; i < 8; ++i) s += p8[i] * pw1[o*8 + i];
  s = (s - pm[o]) * pg[o] * rsqrtf(pv[o] + 1e-5f) + pb[o];
  sh[o] = fmaxf(s, 0.0f);
  __syncthreads();
  float acc = pb2[o];
  for (int k = 0; k < DMODEL; ++k) acc += sh[k] * pw2[o*DMODEL + k];
  float hv = h[(size_t)pos*DMODEL + o];
  hv = (hv - em[o]) * eg[o] * rsqrtf(ev[o] + 1e-3f) + eb[o];
  h[(size_t)pos*DMODEL + o] = fmaxf(hv, 0.0f) + acc;
}

// ---------------------------------------------------------------- per-batch bitonic argsort (8192 keys in LDS)
__global__ void __launch_bounds__(1024) k_argsort(const int* __restrict__ coords,
                                                  const int* __restrict__ curve,
                                                  int* __restrict__ order, int* __restrict__ inv)
{
  __shared__ int key[LSEQ];
  __shared__ int val[LSEQ];
  const int b = blockIdx.x;
  const int tid = threadIdx.x;
  for (int j = tid; j < LSEQ; j += 1024) {
    int idx = b*LSEQ + j;
    int y = coords[idx*3+1], x = coords[idx*3+2];
    key[j] = curve[y*WGRD + x];
    val[j] = j;
  }
  __syncthreads();
  for (int k = 2; k <= LSEQ; k <<= 1) {
    for (int jj = k >> 1; jj > 0; jj >>= 1) {
      for (int t = tid; t < LSEQ/2; t += 1024) {
        int i = ((t / jj) * (jj << 1)) + (t % jj);
        int ixj = i + jj;
        bool up = ((i & k) == 0);
        int a = key[i], c = key[ixj];
        if ((a > c) == up) {
          key[i] = c; key[ixj] = a;
          int tv = val[i]; val[i] = val[ixj]; val[ixj] = tv;
        }
      }
      __syncthreads();
    }
  }
  for (int j = tid; j < LSEQ; j += 1024) {
    order[b*LSEQ + j] = val[j];
    inv[b*LSEQ + val[j]] = j;
  }
}

__global__ void k_gather(const float* __restrict__ h, const int* __restrict__ order,
                         float* __restrict__ hs)
{
  int gid = blockIdx.x * 256 + threadIdx.x;  // NPTS*32
  int row = gid >> 5, q = gid & 31;
  int b = row >> 13;
  int src = (b << 13) + order[row];
  ((float4*)hs)[(size_t)row*32 + q] = ((const float4*)h)[(size_t)src*32 + q];
}

__global__ void k_rms(const float* __restrict__ hs, float* __restrict__ rscale)
{
  int wave = threadIdx.x >> 5, lane = threadIdx.x & 31;
  int row = blockIdx.x * 8 + wave;
  float4 v = *(const float4*)(hs + (size_t)row*DMODEL + lane*4);
  float ss = v.x*v.x + v.y*v.y + v.z*v.z + v.w*v.w;
#pragma unroll
  for (int m = 16; m > 0; m >>= 1) ss += __shfl_xor(ss, m, 32);
  if (lane == 0) rscale[row] = rsqrtf(ss * (1.0f/DMODEL) + 1e-5f);
}

// ---------------------------------------------------------------- depthwise causal conv4 + silu
__global__ void k_dwconv(const float* __restrict__ xz, const float* __restrict__ cw,
                         const float* __restrict__ cb, float* __restrict__ u)
{
  int gid = blockIdx.x * 256 + threadIdx.x;  // NPTS*256
  int row = gid >> 8, d = gid & 255;
  int b = row >> 13, l = row & (LSEQ-1);
  float acc = cb[d];
#pragma unroll
  for (int k = 0; k < 4; ++k) {
    int ll = l - 3 + k;
    if (ll >= 0) acc += xz[(size_t)((b << 13) + ll)*512 + d] * cw[d*4 + k];
  }
  u[gid] = silu(acc);
}

// ---------------------------------------------------------------- dt = softplus(xdbl[:,:8] @ dtw^T + dtb)
__global__ void k_dt(const float* __restrict__ xdbl, const float* __restrict__ dtw,
                     const float* __restrict__ dtb, float* __restrict__ dt)
{
  int gid = blockIdx.x * 256 + threadIdx.x;  // NPTS*256
  int row = gid >> 8, d = gid & 255;
  const float* xr = xdbl + (size_t)row*48;
  float acc = dtb[d];
#pragma unroll
  for (int r = 0; r < 8; ++r) acc += xr[r] * dtw[d*8 + r];
  dt[gid] = (acc > 20.0f) ? acc : log1pf(__expf(acc));
}

// ---------------------------------------------------------------- selective scan
// one wave per workgroup: 32 waves spread over 32 SIMDs, minimizing contention
// on the serial exp/fma dependency chain. B/C streamed through LDS in 128-step
// chunks; epilogue y=(scan+u*D)*silu(z) fused, written in place over u.
__global__ void __launch_bounds__(32) k_scan(
    const float* __restrict__ dt, const float* __restrict__ xdbl,
    const float* __restrict__ xz, const float* __restrict__ Alog,
    const float* __restrict__ Dv, float* __restrict__ u)
{
  __shared__ float sBC[128*32];
  const int b = blockIdx.x;
  const int d = blockIdx.y * 32 + threadIdx.x;
  float Ad[16];
#pragma unroll
  for (int s = 0; s < 16; ++s) Ad[s] = -__expf(Alog[d*16 + s]);
  const float Dd = Dv[d];
  float h[16];
#pragma unroll
  for (int s = 0; s < 16; ++s) h[s] = 0.0f;

  for (int c = 0; c < LSEQ/128; ++c) {
    int rowbase = (b << 13) + c*128;
    for (int i = threadIdx.x; i < 128*32; i += 32) {
      int st = i >> 5, cc = i & 31;
      sBC[i] = xdbl[(size_t)(rowbase + st)*48 + 8 + cc];
    }
    __syncthreads();
    for (int st = 0; st < 128; ++st) {
      int row = rowbase + st;
      if ((st & 15) == 0) {
        // pull next rows of the dependent stream ahead of the serial chain
        __builtin_prefetch(u  + (size_t)(row + 64)*256 + d, 0, 0);
        __builtin_prefetch(dt + (size_t)(row + 64)*256 + d, 0, 0);
        __builtin_prefetch(xz + (size_t)(row + 64)*512 + 256 + d, 0, 0);
      }
      float uu  = u[(size_t)row*256 + d];
      float dtv = dt[(size_t)row*256 + d];
      float du = dtv * uu;
      const float* bc = sBC + st*32;
      float y = 0.0f;
#pragma unroll
      for (int s = 0; s < 16; ++s) {
        float a = __expf(dtv * Ad[s]);
        h[s] = h[s]*a + du*bc[s];
        y += h[s]*bc[16 + s];
      }
      float z = xz[(size_t)row*512 + 256 + d];
      u[(size_t)row*256 + d] = (y + uu*Dd) * silu(z);
    }
    __syncthreads();
  }
}

// ---------------------------------------------------------------- inverse reorder + dual LayerNorm
__global__ void k_scatter_ln(const float* __restrict__ yf, const float* __restrict__ yb,
                             const int* __restrict__ inv,
                             const float* __restrict__ lng, const float* __restrict__ lnb,
                             const float* __restrict__ lbg, const float* __restrict__ lbb,
                             float* __restrict__ outln)
{
  int wave = threadIdx.x >> 5, lane = threadIdx.x & 31;
  int pos = blockIdx.x * 8 + wave;
  int b = pos >> 13;
  int lf = inv[pos];
  int rowf = (b << 13) + lf;
  int rowb = (b << 13) + (LSEQ-1 - lf);
  float4 vf = *(const float4*)(yf + (size_t)rowf*DMODEL + lane*4);
  float4 vb = *(const float4*)(yb + (size_t)rowb*DMODEL + lane*4);
  float sf = vf.x+vf.y+vf.z+vf.w;
  float sb = vb.x+vb.y+vb.z+vb.w;
  float qf = vf.x*vf.x+vf.y*vf.y+vf.z*vf.z+vf.w*vf.w;
  float qb = vb.x*vb.x+vb.y*vb.y+vb.z*vb.z+vb.w*vb.w;
#pragma unroll
  for (int m = 16; m > 0; m >>= 1) {
    sf += __shfl_xor(sf, m, 32); qf += __shfl_xor(qf, m, 32);
    sb += __shfl_xor(sb, m, 32); qb += __shfl_xor(qb, m, 32);
  }
  float muf = sf*(1.0f/DMODEL), mub = sb*(1.0f/DMODEL);
  float rf = rsqrtf(qf*(1.0f/DMODEL) - muf*muf + 1e-5f);
  float rb = rsqrtf(qb*(1.0f/DMODEL) - mub*mub + 1e-5f);
  int k = lane*4;
  float4 o;
  o.x = (vf.x-muf)*rf*lng[k+0]+lnb[k+0] + (vb.x-mub)*rb*lbg[k+0]+lbb[k+0];
  o.y = (vf.y-muf)*rf*lng[k+1]+lnb[k+1] + (vb.y-mub)*rb*lbg[k+1]+lbb[k+1];
  o.z = (vf.z-muf)*rf*lng[k+2]+lnb[k+2] + (vb.z-mub)*rb*lbg[k+2]+lbb[k+2];
  o.w = (vf.w-muf)*rf*lng[k+3]+lnb[k+3] + (vb.w-mub)*rb*lbg[k+3]+lbb[k+3];
  *(float4*)(outln + (size_t)pos*DMODEL + k) = o;
}

// ---------------------------------------------------------------- decoder BN epilogue
__global__ void k_final(const float* __restrict__ dec, const float* __restrict__ feats,
                        const float* g1, const float* b1, const float* m1, const float* v1,
                        const float* g2, const float* b2, const float* m2, const float* v2,
                        const float* go, const float* bo, const float* mo, const float* vo,
                        float* __restrict__ out)
{
  int gid = blockIdx.x * 256 + threadIdx.x;
  int c = gid & 127;
  float x = dec[gid];
  x = (x - m1[c]) * g1[c] * rsqrtf(v1[c] + 1e-3f) + b1[c];
  x = fmaxf(x, 0.0f);
  x = (x - m2[c]) * g2[c] * rsqrtf(v2[c] + 1e-3f) + b2[c];
  x = feats[gid] + x;
  x = (x - mo[c]) * go[c] * rsqrtf(vo[c] + 1e-3f) + bo[c];
  out[gid] = x;
}

// ================================================================ host
struct MParams {
  const float *norm, *in_proj, *conv_w, *conv_b, *xproj, *dtw, *dtb, *A_log, *D, *outproj;
};

extern "C" void kernel_launch(void* const* d_in, const int* in_sizes, int n_in,
                              void* d_out, int out_size, void* d_ws, size_t ws_size,
                              hipStream_t stream)
{
  (void)in_sizes; (void)n_in; (void)out_size; (void)ws_size;

  const float* feats  = (const float*)d_in[0];
  const int*   coords = (const int*)d_in[1];
  const int*   curve  = (const int*)d_in[2];
  // d_in[3] = batch_size (compile-time NBATCH)
  const float* enc_w  = (const float*)d_in[4];
  const float* dec_w  = (const float*)d_in[5];
  const float* pos_w1 = (const float*)d_in[6];
  const float* pos_b1 = (const float*)d_in[7];
  const float* pos_w2 = (const float*)d_in[8];
  const float* pos_b2 = (const float*)d_in[9];
  const float* enc_bn[4]; const float* pos_bn[4]; const float* dec_bn1[4];
  const float* dec_bn2[4]; const float* out_bn[4];
  for (int i = 0; i < 4; ++i) {
    enc_bn[i]  = (const float*)d_in[10 + i];
    pos_bn[i]  = (const float*)d_in[14 + i];
    dec_bn1[i] = (const float*)d_in[18 + i];
    dec_bn2[i] = (const float*)d_in[22 + i];
    out_bn[i]  = (const float*)d_in[26 + i];
  }
  MParams mp[2];
  for (int m = 0; m < 2; ++m) {
    int base = 30 + 10*m;
    mp[m].norm    = (const float*)d_in[base + 0];
    mp[m].in_proj = (const float*)d_in[base + 1];
    mp[m].conv_w  = (const float*)d_in[base + 2];
    mp[m].conv_b  = (const float*)d_in[base + 3];
    mp[m].xproj   = (const float*)d_in[base + 4];
    mp[m].dtw     = (const float*)d_in[base + 5];
    mp[m].dtb     = (const float*)d_in[base + 6];
    mp[m].A_log   = (const float*)d_in[base + 7];
    mp[m].D       = (const float*)d_in[base + 8];
    mp[m].outproj = (const float*)d_in[base + 9];
  }
  const float* ln_g  = (const float*)d_in[50];
  const float* ln_b  = (const float*)d_in[51];
  const float* lnb_g = (const float*)d_in[52];
  const float* lnb_b = (const float*)d_in[53];

  // ---- workspace bump allocator
  char* wsp = (char*)d_ws;
  auto alloc = [&](size_t bytes) -> void* {
    void* p = (void*)wsp;
    wsp += (bytes + 255) & ~(size_t)255;
    return p;
  };
  int*   vol    = (int*)  alloc((size_t)NBATCH*HGRD*WGRD*4);
  int*   order  = (int*)  alloc((size_t)NPTS*4);
  int*   inv    = (int*)  alloc((size_t)NPTS*4);
  float* h      = (float*)alloc((size_t)NPTS*DMODEL*4);   // enc conv out -> h; reused for dec conv out
  float* hs     = (float*)alloc((size_t)NPTS*DMODEL*4);
  float* rscale = (float*)alloc((size_t)NPTS*4);
  float* xz     = (float*)alloc((size_t)NPTS*512*4);
  float* ubuf   = (float*)alloc((size_t)NPTS*256*4);      // conv/silu out -> scan result in place
  float* xdbl   = (float*)alloc((size_t)NPTS*48*4);
  float* dtbuf  = (float*)alloc((size_t)NPTS*256*4);
  float* yf     = (float*)alloc((size_t)NPTS*DMODEL*4);
  float* yb     = (float*)alloc((size_t)NPTS*DMODEL*4);
  float* outln  = (float*)alloc((size_t)NPTS*DMODEL*4);
  unsigned short* encWb = (unsigned short*)alloc((size_t)9*128*128*2);
  unsigned short* decWb = (unsigned short*)alloc((size_t)9*128*128*2);
  unsigned short* inW[2]  = { (unsigned short*)alloc((size_t)512*128*2),
                              (unsigned short*)alloc((size_t)512*128*2) };
  unsigned short* xpW[2]  = { (unsigned short*)alloc((size_t)48*256*2),
                              (unsigned short*)alloc((size_t)48*256*2) };
  unsigned short* opW[2]  = { (unsigned short*)alloc((size_t)128*256*2),
                              (unsigned short*)alloc((size_t)128*256*2) };

  // ---- prep
  k_init_vol<<<NBATCH*HGRD*WGRD/256, 256, 0, stream>>>(vol);
  k_scatter_vol<<<NPTS/256, 256, 0, stream>>>(coords, vol);
  k_convw2bf<<<9*128*128/256, 256, 0, stream>>>(enc_w, encWb);
  k_convw2bf<<<9*128*128/256, 256, 0, stream>>>(dec_w, decWb);
  for (int m = 0; m < 2; ++m) {
    k_f2bf<<<512*128/256, 256, 0, stream>>>(mp[m].in_proj, inW[m]);
    k_f2bf_pad<<<48*256/256, 256, 0, stream>>>(mp[m].xproj, xpW[m]);
    k_f2bf<<<128*256/256, 256, 0, stream>>>(mp[m].outproj, opW[m]);
  }

  // ---- encoder conv + BN/ReLU + positional MLP
  k_conv<<<dim3(NPTS/64, 2), 128, 0, stream>>>(feats, coords, vol, encWb, h);
  k_pe_add<<<NPTS, 128, 0, stream>>>(h, coords, pos_w1, pos_b1, pos_w2, pos_b2,
                                     pos_bn[0], pos_bn[1], pos_bn[2], pos_bn[3],
                                     enc_bn[0], enc_bn[1], enc_bn[2], enc_bn[3]);

  // ---- curve reorder
  k_argsort<<<NBATCH, 1024, 0, stream>>>(coords, curve, order, inv);
  k_gather<<<NPTS*32/256, 256, 0, stream>>>(h, order, hs);
  k_rms<<<NPTS/8, 256, 0, stream>>>(hs, rscale);

  // ---- two Mamba branches (m=0 forward, m=1 backward via flipped coords)
  for (int m = 0; m < 2; ++m) {
    float* ydst = (m == 0) ? yf : yb;
    k_gemm<8,128><<<dim3(NPTS/64, 4), 128, 0, stream>>>(hs, DMODEL, rscale, mp[m].norm,
                                                        inW[m], xz, 512, m);
    k_dwconv<<<NPTS*256/256, 256, 0, stream>>>(xz, mp[m].conv_w, mp[m].conv_b, ubuf);
    k_gemm<3,256><<<dim3(NPTS/64, 1), 128, 0, stream>>>(ubuf, 256, nullptr, nullptr,
                                                        xpW[m], xdbl, 48, 0);
    k_dt<<<NPTS*256/256, 256, 0, stream>>>(xdbl, mp[m].dtw, mp[m].dtb, dtbuf);
    k_scan<<<dim3(NBATCH, 8), 32, 0, stream>>>(dtbuf, xdbl, xz, mp[m].A_log, mp[m].D, ubuf);
    k_gemm<8,256><<<dim3(NPTS/64, 1), 128, 0, stream>>>(ubuf, 256, nullptr, nullptr,
                                                        opW[m], ydst, DMODEL, 0);
  }

  // ---- inverse reorder + dual LN, decoder conv, BN epilogue
  k_scatter_ln<<<NPTS/8, 256, 0, stream>>>(yf, yb, inv, ln_g, ln_b, lnb_g, lnb_b, outln);
  k_conv<<<dim3(NPTS/64, 2), 128, 0, stream>>>(outln, coords, vol, decWb, h);
  k_final<<<NPTS*128/256, 256, 0, stream>>>(h, feats,
                                            dec_bn1[0], dec_bn1[1], dec_bn1[2], dec_bn1[3],
                                            dec_bn2[0], dec_bn2[1], dec_bn2[2], dec_bn2[3],
                                            out_bn[0], out_bn[1], out_bn[2], out_bn[3],
                                            (float*)d_out);
}
